// SSIMLoss_57337813401724
// MI455X (gfx1250) — compile-verified, matrix operand order checked by reading, not run on hardware
//
#include <hip/hip_runtime.h>
#include <math.h>

typedef __attribute__((ext_vector_type(16))) _Float16 v16h;
typedef __attribute__((ext_vector_type(8)))  float    v8f;

#define IMG_H 512
#define IMG_W 512
#define NPLANES 48                 // 16 * 3
#define TILES_PER_PLANE 1024       // 32 * 32 tiles of 16x16
#define TOTAL_TILES (NPLANES * TILES_PER_PLANE)   // 49152
#define WAVES_PER_BLOCK 4
#define BLOCK_THREADS (WAVES_PER_BLOCK * 32)      // 128
#define NBLOCKS (TOTAL_TILES / WAVES_PER_BLOCK)   // 12288
#define TOTAL_ELEMS 12582912.0
#define LDS_STRIDE 33              // 32 + 1 pad: avoids 64-bank conflicts

__device__ __forceinline__ v8f wmma16(v16h a, v16h b, v8f c) {
    // D = A(16x32 f16) * B(32x16 f16) + C(16x16 f32)
    return __builtin_amdgcn_wmma_f32_16x16x32_f16(
        /*neg_a=*/false, a, /*neg_b=*/false, b,
        /*c_mod=*/(short)0, c, /*reuse_a=*/false, /*reuse_b=*/false);
}

__global__ __launch_bounds__(BLOCK_THREADS)
void ssim_tile_kernel(const float* __restrict__ img,
                      const float* __restrict__ tgt,
                      float* __restrict__ partial)
{
    __shared__ float lds_x[WAVES_PER_BLOCK][32 * LDS_STRIDE];
    __shared__ float lds_y[WAVES_PER_BLOCK][32 * LDS_STRIDE];

    const int lane   = threadIdx.x & 31;
    const int waveId = threadIdx.x >> 5;
    const int tile   = blockIdx.x * WAVES_PER_BLOCK + waveId;   // < TOTAL_TILES exactly

    const int plane = tile >> 10;            // / 1024
    const int tIdx  = tile & 1023;
    const int r0 = (tIdx >> 5) << 4;         // tile row origin
    const int c0 = (tIdx & 31) << 4;         // tile col origin

    const float* __restrict__ ip = img + (size_t)plane * (IMG_H * IMG_W);
    const float* __restrict__ tp = tgt + (size_t)plane * (IMG_H * IMG_W);

    // ---- Stage haloed 32x32 region into this wave's private LDS slice.
    // lane = column -> each global_load_b32 is one fully-coalesced 128B line.
    // Clamped addresses keep every lane in-bounds (coalescing preserved);
    // out-of-image texels become 0.0f = reference's zero padding, and all
    // staged values are finite (no 0*NaN hazard inside WMMA).
    {
        float* sx = &lds_x[waveId][0];
        float* sy = &lds_y[waveId][0];
        const int c  = c0 - 5 + lane;
        const int cc = min(max(c, 0), IMG_W - 1);
        const bool cin = ((unsigned)c < (unsigned)IMG_W);
        #pragma unroll
        for (int i = 0; i < 32; ++i) {
            int r = r0 - 5 + i;
            int rc = min(max(r, 0), IMG_H - 1);
            bool inb = cin && ((unsigned)r < (unsigned)IMG_H);
            size_t off = (size_t)rc * IMG_W + cc;
            float vx = ip[off];
            float vy = tp[off];
            sx[i * LDS_STRIDE + lane] = inb ? vx : 0.0f;
            sy[i * LDS_STRIDE + lane] = inb ? vy : 0.0f;
        }
    }

    const int half  = lane >> 4;             // 0 = lanes 0-15, 1 = lanes 16-31
    const int rl    = lane & 15;
    const int kbase = half * 8;

    // ---- Gaussian weights g[0..10], sigma = 1.5, normalized (f32) ----
    float g[11];
    {
        float s = 0.0f;
        #pragma unroll
        for (int i = 0; i < 11; ++i) {
            float d = (float)(i - 5);
            g[i] = expf(-d * d / 4.5f);
            s += g[i];
        }
        float inv = 1.0f / s;
        #pragma unroll
        for (int i = 0; i < 11; ++i) g[i] *= inv;
    }

    // ---- Banded weight fragment: value(e) = g[K(e) - (lane&15)] within band.
    // Identical per-lane values serve as B of the horizontal pass (g[k-n])
    // and as A of the vertical pass (g[k-m]).  K(e) per ISA 16-bit A/B layout.
    v16h wfrag;
    #pragma unroll
    for (int e = 0; e < 16; ++e) {
        int K = (e < 8) ? (kbase + e) : (16 + kbase + (e - 8));
        int d = K - rl;
        wfrag[e] = (d >= 0 && d <= 10) ? (_Float16)g[d] : (_Float16)0.0f;
    }

    // ---- Gather A-fragment values from LDS (same-wave, DScnt-ordered).
    // A-fragment row-group grp: region row = grp*16 + (lane&15), col = K(e).
    float xv[2][16], yv[2][16];
    {
        const float* sx = &lds_x[waveId][0];
        const float* sy = &lds_y[waveId][0];
        #pragma unroll
        for (int grp = 0; grp < 2; ++grp) {
            int rr = grp * 16 + rl;
            #pragma unroll
            for (int e = 0; e < 16; ++e) {
                int K = (e < 8) ? (kbase + e) : (16 + kbase + (e - 8));
                xv[grp][e] = sx[rr * LDS_STRIDE + K];
                yv[grp][e] = sy[rr * LDS_STRIDE + K];
            }
        }
    }

    // ---- Per field: horizontal blur (2 WMMAs over row-groups), then the
    // f32 D fragments convert lane-locally into the f16 B fragment of the
    // vertical pass (D0[e] -> B[e], D1[e] -> B[8+e]).
    v16h bv[5];
    #pragma unroll
    for (int f = 0; f < 5; ++f) {
        v16h a0, a1;
        #pragma unroll
        for (int e = 0; e < 16; ++e) {
            float p0, p1;
            switch (f) {
                case 0:  p0 = xv[0][e];            p1 = xv[1][e];            break;
                case 1:  p0 = yv[0][e];            p1 = yv[1][e];            break;
                case 2:  p0 = xv[0][e] * xv[0][e]; p1 = xv[1][e] * xv[1][e]; break;
                case 3:  p0 = yv[0][e] * yv[0][e]; p1 = yv[1][e] * yv[1][e]; break;
                default: p0 = xv[0][e] * yv[0][e]; p1 = xv[1][e] * yv[1][e]; break;
            }
            a0[e] = (_Float16)p0;
            a1[e] = (_Float16)p1;
        }
        v8f z = {};
        v8f d0 = wmma16(a0, wfrag, z);   // hb rows 0..15 of region
        v8f d1 = wmma16(a1, wfrag, z);   // hb rows 16..31 (26..31 unused downstream)
        v16h b;
        #pragma unroll
        for (int e = 0; e < 8; ++e) {
            b[e]     = (_Float16)d0[e];
            b[8 + e] = (_Float16)d1[e];
        }
        bv[f] = b;
    }

    // ---- Vertical blur: Out = Wfrag(16x32) * Hb(32x16) ----
    v8f acc[5];
    #pragma unroll
    for (int f = 0; f < 5; ++f) {
        v8f z = {};
        acc[f] = wmma16(wfrag, bv[f], z);
    }

    // ---- SSIM map on the 8 D elements this lane owns, sum them ----
    const float C1 = 1e-4f;   // 0.01^2
    const float C2 = 9e-4f;   // 0.03^2
    float lsum = 0.0f;
    #pragma unroll
    for (int e = 0; e < 8; ++e) {
        float m1  = acc[0][e], m2 = acc[1][e];
        float bxx = acc[2][e], byy = acc[3][e], bxy = acc[4][e];
        float m1s = m1 * m1, m2s = m2 * m2, m12 = m1 * m2;
        float s1  = bxx - m1s, s2 = byy - m2s, s12 = bxy - m12;
        float num = (2.0f * m12 + C1) * (2.0f * s12 + C2);
        float den = (m1s + m2s + C1) * (s1 + s2 + C2);
        lsum += num / den;
    }

    // ---- wave32 reduction, then per-block partial (deterministic order) ----
    #pragma unroll
    for (int off = 16; off > 0; off >>= 1)
        lsum += __shfl_xor(lsum, off, 32);

    __shared__ float wsum[WAVES_PER_BLOCK];
    if (lane == 0) wsum[waveId] = lsum;
    __syncthreads();
    if (threadIdx.x == 0) {
        float b = 0.0f;
        #pragma unroll
        for (int i = 0; i < WAVES_PER_BLOCK; ++i) b += wsum[i];
        partial[blockIdx.x] = b;
    }
}

__global__ __launch_bounds__(256)
void ssim_reduce_kernel(const float* __restrict__ partial, float* __restrict__ out)
{
    __shared__ double sh[256];
    double s = 0.0;
    for (int i = threadIdx.x; i < NBLOCKS; i += 256)
        s += (double)partial[i];
    sh[threadIdx.x] = s;
    __syncthreads();
    for (int off = 128; off > 0; off >>= 1) {
        if (threadIdx.x < off) sh[threadIdx.x] += sh[threadIdx.x + off];
        __syncthreads();
    }
    if (threadIdx.x == 0)
        out[0] = (float)(1.0 - sh[0] / TOTAL_ELEMS);
}

extern "C" void kernel_launch(void* const* d_in, const int* in_sizes, int n_in,
                              void* d_out, int out_size, void* d_ws, size_t ws_size,
                              hipStream_t stream) {
    const float* img = (const float*)d_in[0];
    const float* tgt = (const float*)d_in[1];
    float* out     = (float*)d_out;
    float* partial = (float*)d_ws;   // NBLOCKS * 4 = 48 KB of scratch

    ssim_tile_kernel<<<NBLOCKS, BLOCK_THREADS, 0, stream>>>(img, tgt, partial);
    ssim_reduce_kernel<<<1, 256, 0, stream>>>(partial, out);
}
